// TopKBalancedNoisyGate_72636486910598
// MI455X (gfx1250) — matrix operand order repaired and checked by reading, not compiled, not run on hardware
//
#include <hip/hip_runtime.h>
#include <hip/hip_bf16.h>
#include <math.h>

typedef __bf16 bf16_t;
typedef __attribute__((ext_vector_type(16))) __bf16 v16bf;
typedef __attribute__((ext_vector_type(8)))  float  v8f;
typedef __attribute__((ext_vector_type(4)))  float  f32x4;
typedef __attribute__((ext_vector_type(4)))  unsigned int u32x4;

#define N_TOK   16384
#define D_MODEL 4096
#define N_EXP   16
#define N_SEL   4
#define NOISE_EPS   0.01f
#define LOSS_LAMBDA 0.01f

#define WAVES_PER_BLOCK 8
#define ROWP 4104                               // padded bf16 row (4096 + 8) -> bank-staggered
#define W_BYTES (N_EXP * ROWP * 2)              // 131328 B per weight matrix
#define LDS_W1 0
#define LDS_WN (W_BYTES)
#define LDS_W2 (2 * W_BYTES)                    // 262656: w2 as f32[256]
#define LDS_SCRATCH (2 * W_BYTES + 1024)        // 263680
#define SCRATCH_PER_WAVE 2048                   // th[16][16] f32 + nc[16][16] f32
#define LDS_TOTAL (LDS_SCRATCH + WAVES_PER_BLOCK * SCRATCH_PER_WAVE)  // 280064 < 320KB

#define PF_DIST 1024                            // prefetch 4KB (~32 k-steps) ahead

__global__ void moe_gate_init_ws(float* ws) {
    if (threadIdx.x < 32) ws[threadIdx.x] = 0.0f;
}

__global__ __launch_bounds__(256)
void moe_gate_main(const float* __restrict__ x,
                   const float* __restrict__ w1,
                   const float* __restrict__ w2,
                   const float* __restrict__ wn,
                   const float* __restrict__ noise,
                   int*   __restrict__ out_idx,
                   float* __restrict__ out_scr,
                   float* __restrict__ ws_imp,
                   float* __restrict__ ws_load) {
    extern __shared__ char smem[];
    bf16_t* w1l = (bf16_t*)(smem + LDS_W1);
    bf16_t* wnl = (bf16_t*)(smem + LDS_WN);
    float*  w2l = (float*)(smem + LDS_W2);

    const int tid = threadIdx.x;

    // ---- Stage w1 / wn into LDS as bf16 (padded rows), w2 as f32 ----
    for (int i = tid; i < (N_EXP * D_MODEL) / 4; i += 256) {
        const int idx = i * 4;
        const int n = idx >> 12;          // / 4096
        const int k = idx & (D_MODEL - 1);
        const f32x4 a = *(const f32x4*)(w1 + idx);
        const f32x4 b = *(const f32x4*)(wn + idx);
        bf16_t* p = w1l + n * ROWP + k;
        p[0] = (bf16_t)a[0]; p[1] = (bf16_t)a[1]; p[2] = (bf16_t)a[2]; p[3] = (bf16_t)a[3];
        bf16_t* q = wnl + n * ROWP + k;
        q[0] = (bf16_t)b[0]; q[1] = (bf16_t)b[1]; q[2] = (bf16_t)b[2]; q[3] = (bf16_t)b[3];
    }
    w2l[tid] = w2[tid];                   // 256 floats
    __syncthreads();

    const int wave = tid >> 5;
    const int lane = tid & 31;
    const int hf   = lane >> 4;           // half-wave select
    const int mn   = lane & 15;           // A: token row m ; B: expert col n
    const int tok_base = (blockIdx.x * WAVES_PER_BLOCK + wave) * 16;

    // ---- K loop: dual bf16 WMMA (h = x*w1^T, g = x*wn^T), f32 accumulate ----
    v8f c_h = {};
    v8f c_n = {};
    const float*  xrow  = x + (size_t)(tok_base + mn) * D_MODEL;
    const bf16_t* b1row = w1l + mn * ROWP;
    const bf16_t* bnrow = wnl + mn * ROWP;

#pragma unroll 4
    for (int kb = 0; kb < D_MODEL; kb += 32) {
        __builtin_prefetch(xrow + kb + PF_DIST, 0, 1);   // global_prefetch_b8 (no LOADcnt)
        // A fragment (16-bit A 16x32 layout): lane holds k in {8h..8h+7, 16+8h..16+8h+7}
        const f32x4 f0 = *(const f32x4*)(xrow + kb + 8 * hf);
        const f32x4 f1 = *(const f32x4*)(xrow + kb + 8 * hf + 4);
        const f32x4 f2 = *(const f32x4*)(xrow + kb + 16 + 8 * hf);
        const f32x4 f3 = *(const f32x4*)(xrow + kb + 16 + 8 * hf + 4);
        v16bf a;
        a[0]=(bf16_t)f0[0]; a[1]=(bf16_t)f0[1]; a[2]=(bf16_t)f0[2]; a[3]=(bf16_t)f0[3];
        a[4]=(bf16_t)f1[0]; a[5]=(bf16_t)f1[1]; a[6]=(bf16_t)f1[2]; a[7]=(bf16_t)f1[3];
        a[8]=(bf16_t)f2[0]; a[9]=(bf16_t)f2[1]; a[10]=(bf16_t)f2[2]; a[11]=(bf16_t)f2[3];
        a[12]=(bf16_t)f3[0]; a[13]=(bf16_t)f3[1]; a[14]=(bf16_t)f3[2]; a[15]=(bf16_t)f3[3];
        // B fragments (16-bit B 32x16): lane holds k in {8h..8h+7, 16+8h..16+8h+7}
        union { u32x4 u[2]; v16bf v; } B1, B2;
        const u32x4* pb1 = (const u32x4*)(b1row + kb + 8 * hf);
        B1.u[0] = pb1[0]; B1.u[1] = pb1[2];
        const u32x4* pb2 = (const u32x4*)(bnrow + kb + 8 * hf);
        B2.u[0] = pb2[0]; B2.u[1] = pb2[2];
        c_h = __builtin_amdgcn_wmma_f32_16x16x32_bf16(false, a, false, B1.v, (short)0, c_h, false, false);
        c_n = __builtin_amdgcn_wmma_f32_16x16x32_bf16(false, a, false, B2.v, (short)0, c_n, false, false);
    }

    // ---- elementwise: tanh(h), softplus(g)+eps; dump tile to LDS token-major ----
    float* thS = (float*)(smem + LDS_SCRATCH + wave * SCRATCH_PER_WAVE);
    float* ncS = thS + 256;
#pragma unroll
    for (int r = 0; r < 8; ++r) {
        const int m = r + 8 * hf;         // token row within tile (C-layout)
        thS[m * 16 + mn] = tanhf(c_h[r]);
        const float v = c_n[r];
        ncS[m * 16 + mn] = (v > 15.0f ? v : log1pf(expf(v))) + NOISE_EPS;
    }
    __syncthreads();

    // ---- per-token epilogue: lanes 0..15 each own one token ----
    if (lane < 16) {
        const int t = tok_base + lane;
        float thv[16], ncl[16], lg[16], lno[16], logit[16];
        const f32x4* nzp = (const f32x4*)(noise + (size_t)t * 16);
        f32x4 nz[4] = { nzp[0], nzp[1], nzp[2], nzp[3] };
#pragma unroll
        for (int e = 0; e < 16; ++e) {
            thv[e] = thS[lane * 16 + e];
            ncl[e] = ncS[lane * 16 + e];
        }
#pragma unroll
        for (int f = 0; f < 16; ++f) {
            float s = 0.0f;
#pragma unroll
            for (int e = 0; e < 16; ++e) s += thv[e] * w2l[f * 16 + e];
            lg[f]    = s;
            lno[f]   = nz[f >> 2][f & 3] * ncl[f];
            logit[f] = lg[f] + lno[f];
        }
        // top-5 of 16, fully unrolled selection (first-max tie-break like jax)
        float v[16];
#pragma unroll
        for (int i = 0; i < 16; ++i) v[i] = logit[i];
        float bv[5]; int bi[5];
#pragma unroll
        for (int j = 0; j < 5; ++j) {
            float mv = v[0]; int mi = 0;
#pragma unroll
            for (int i = 1; i < 16; ++i) { if (v[i] > mv) { mv = v[i]; mi = i; } }
            bv[j] = mv; bi[j] = mi;
#pragma unroll
            for (int i = 0; i < 16; ++i) { if (i == mi) v[i] = -3.4e38f; }
        }
        // softmax over top-4 (bv[0] is the max)
        const float s1 = expf(bv[1] - bv[0]);
        const float s2 = expf(bv[2] - bv[0]);
        const float s3 = expf(bv[3] - bv[0]);
        const float inv = 1.0f / (1.0f + s1 + s2 + s3);
        const float sc[4] = { inv, s1 * inv, s2 * inv, s3 * inv };
#pragma unroll
        for (int j = 0; j < N_SEL; ++j) {
            out_idx[t * N_SEL + j] = bi[j];
            out_scr[t * N_SEL + j] = sc[j];
        }
        // load probabilities (normal CDF) + importance scatter
        const float thr_in = bv[4], thr_out = bv[3];
        const float INV_SQRT2 = 0.70710678118654752440f;
        float imp[16], ldp[16];
#pragma unroll
        for (int e = 0; e < 16; ++e) {
            const bool  is_in = lno[e] > thr_in;
            const float thr   = is_in ? thr_in : thr_out;
            const float z     = (lg[e] - thr) / ncl[e];
            ldp[e] = 0.5f * erfcf(-z * INV_SQRT2);
            float s = 0.0f;
#pragma unroll
            for (int j = 0; j < N_SEL; ++j) s += (bi[j] == e) ? sc[j] : 0.0f;
            imp[e] = s;
        }
        // reduce across the 16 token-lanes, one atomic per expert per wave
#pragma unroll
        for (int e = 0; e < 16; ++e) {
            float a = imp[e], b = ldp[e];
#pragma unroll
            for (int off = 8; off >= 1; off >>= 1) {
                a += __shfl_xor(a, off, 32);
                b += __shfl_xor(b, off, 32);
            }
            if (lane == 0) {
                atomicAdd(&ws_imp[e],  a);
                atomicAdd(&ws_load[e], b);
            }
        }
    }
}

__global__ void moe_gate_loss(const float* __restrict__ ws, float* __restrict__ out_loss) {
    if (threadIdx.x == 0 && blockIdx.x == 0) {
        float loss = 0.0f;
        for (int h = 0; h < 2; ++h) {
            const float* v = ws + h * 16;
            float m = 0.0f;
            for (int i = 0; i < 16; ++i) m += v[i];
            m *= (1.0f / 16.0f);
            float var = 0.0f;
            for (int i = 0; i < 16; ++i) { const float d = v[i] - m; var += d * d; }
            var *= (1.0f / 15.0f);                 // ddof=1
            loss += var / (m * m + 1e-10f);
        }
        out_loss[0] = LOSS_LAMBDA * loss;
    }
}

extern "C" void kernel_launch(void* const* d_in, const int* in_sizes, int n_in,
                              void* d_out, int out_size, void* d_ws, size_t ws_size,
                              hipStream_t stream) {
    const float* x     = (const float*)d_in[0];
    const float* w1    = (const float*)d_in[1];
    const float* w2    = (const float*)d_in[2];
    const float* wn    = (const float*)d_in[3];
    const float* noise = (const float*)d_in[4];

    int*   out_idx  = (int*)d_out;                                  // 16384*4 int32
    float* out_scr  = (float*)d_out + N_TOK * N_SEL;                // 16384*4 f32
    float* out_loss = (float*)d_out + 2 * N_TOK * N_SEL;            // 1 f32
    float* ws       = (float*)d_ws;                                 // [0..15]=importance, [16..31]=load

    hipLaunchKernelGGL(moe_gate_init_ws, dim3(1), dim3(32), 0, stream, ws);
    const dim3 grid(N_TOK / (WAVES_PER_BLOCK * 16));                // 128 blocks
    hipLaunchKernelGGL(moe_gate_main, grid, dim3(256), LDS_TOTAL, stream,
                       x, w1, w2, wn, noise, out_idx, out_scr, ws, ws + 16);
    hipLaunchKernelGGL(moe_gate_loss, dim3(1), dim3(32), 0, stream, ws, out_loss);
}
